// Upsample_54752243089544
// MI455X (gfx1250) — matrix operand order repaired and checked by reading, not compiled, not run on hardware
//
#include <hip/hip_runtime.h>
#include <hip/hip_bf16.h>

// ---------------------------------------------------------------------------
// Upsample(2x, FIR [1,3,3,1]) + 3x3 conv + bias for MI455X (gfx1250, wave32).
// Stage 1: h = upfirdn2d(x) in f16 -> d_ws          (bandwidth bound, ~200MB)
// Stage 2: W f32 -> f16 -> d_ws                     (tiny)
// Stage 3: implicit-GEMM 3x3 conv via v_wmma_f32_16x16x32_f16
//          M=Cout(256), N=spatial(128x128 per batch), K=Cin*9=2304
//          Software-pipelined + double-buffered LDS (1 barrier / K-chunk)
//          + sched_group_barrier pipeline: ds_load / wmma lookahead-2.
// ---------------------------------------------------------------------------

typedef __attribute__((ext_vector_type(16))) _Float16 v16h;
typedef __attribute__((ext_vector_type(8)))  _Float16 v8h;
typedef __attribute__((ext_vector_type(8)))  float    v8f;

#define B_    16
#define C_    256
#define HIN   64
#define HOUT  128
#define KTOT  2304      // 256*9
#define KCH   32        // WMMA K chunk
#define MTILE 128       // Cout per WG
#define NT_R  8         // spatial rows per WG tile
#define NT_C  16        // spatial cols per WG tile  (NTILE = 128)
#define LDSP  40        // LDS row pitch in halfs (80B: 16B-aligned, conflict-free)
#define ASZ   (MTILE * LDSP)
#define BSZ   (128 * LDSP)

#if defined(__has_builtin)
#  if __has_builtin(__builtin_amdgcn_sched_group_barrier)
#    define SCHED_GB(mask, cnt, sid) __builtin_amdgcn_sched_group_barrier(mask, cnt, sid)
#  endif
#endif
#ifndef SCHED_GB
#  define SCHED_GB(mask, cnt, sid)
#endif

// ---------------------------------------------------------------------------
// Kernel 1: 2x upsample with separable FIR.  Per-axis taps for phase:
//   even output: (prev, cur) weights (1/4, 3/4); odd: (cur, next) (3/4, 1/4).
// ---------------------------------------------------------------------------
__global__ __launch_bounds__(256) void upfir_kernel(
    const float* __restrict__ x, _Float16* __restrict__ h) {
  size_t idx = (size_t)blockIdx.x * 256 + threadIdx.x;   // over 16*256*128*128
  int ox = (int)(idx & 127);
  int oy = (int)((idx >> 7) & 127);
  size_t bc = idx >> 14;                                 // b*256 + c
  const float* xp = x + bc * (HIN * HIN);

  int iy0, ix0; float wy0, wx0;
  if (oy & 1) { iy0 = (oy - 1) >> 1; wy0 = 0.75f; }
  else        { iy0 = (oy >> 1) - 1; wy0 = 0.25f; }
  if (ox & 1) { ix0 = (ox - 1) >> 1; wx0 = 0.75f; }
  else        { ix0 = (ox >> 1) - 1; wx0 = 0.25f; }
  int iy1 = iy0 + 1, ix1 = ix0 + 1;
  float wy1 = 1.0f - wy0, wx1 = 1.0f - wx0;

  bool x0ok = (unsigned)ix0 < (unsigned)HIN;
  bool x1ok = (unsigned)ix1 < (unsigned)HIN;
  float r0 = 0.0f, r1 = 0.0f;
  if ((unsigned)iy0 < (unsigned)HIN) {
    const float* row = xp + iy0 * HIN;
    r0 = (x0ok ? row[ix0] * wx0 : 0.0f) + (x1ok ? row[ix1] * wx1 : 0.0f);
  }
  if ((unsigned)iy1 < (unsigned)HIN) {
    const float* row = xp + iy1 * HIN;
    r1 = (x0ok ? row[ix0] * wx0 : 0.0f) + (x1ok ? row[ix1] * wx1 : 0.0f);
  }
  h[idx] = (_Float16)(r0 * wy0 + r1 * wy1);
}

// ---------------------------------------------------------------------------
// Kernel 2: weight f32 -> f16 (layout [Cout][Cin*9] is W's natural flatten)
// ---------------------------------------------------------------------------
__global__ __launch_bounds__(256) void wconv_kernel(
    const float* __restrict__ W, _Float16* __restrict__ Wh, int n) {
  int i = blockIdx.x * 256 + threadIdx.x;
  if (i < n) Wh[i] = (_Float16)W[i];
}

// ---------------------------------------------------------------------------
// Kernel 3: implicit GEMM 3x3 conv, software pipelined, double-buffered LDS.
// grid: x = ox-tile (8), y = oy-tile (16), z = batch*2 + coutTile (32)
// block: 256 threads = 8 wave32; wave w computes M rows [16w,16w+16) x N=128.
// B staging: thread owns fixed k (=t&31) and 16 contiguous columns of row
// nb (=t>>5): (ci,kh,kw) decoded once per K-chunk, loads are contiguous.
// ---------------------------------------------------------------------------
__global__ __launch_bounds__(256) void conv3x3_wmma_kernel(
    const _Float16* __restrict__ h, const _Float16* __restrict__ Wh,
    const float* __restrict__ bias, float* __restrict__ y) {

  const int oxBase   = blockIdx.x * NT_C;
  const int oyBase   = blockIdx.y * NT_R;
  const int batch    = blockIdx.z >> 1;
  const int coutBase = (blockIdx.z & 1) * MTILE;

  const int t     = threadIdx.x;
  const int wave  = t >> 5;
  const int lane  = t & 31;
  const int lhalf = lane & 15;      // N / M index inside a fragment
  const int hi    = lane >> 4;      // K-half selector per ISA layout

  __shared__ _Float16 At[2 * ASZ];  // [buf][m][k] weights chunk
  __shared__ _Float16 Bt[2 * BSZ];  // [buf][n][k] im2col chunk

  v8f acc[8];
  #pragma unroll
  for (int i = 0; i < 8; ++i)
    #pragma unroll
    for (int j = 0; j < 8; ++j) acc[i][j] = 0.0f;

  const _Float16* hB = h + (size_t)batch * C_ * HOUT * HOUT;

  // --- A staging: thread t -> row m = t/2, half-range ak = (t&1)*16 ---
  const int am = t >> 1;
  const int ak = (t & 1) * 16;
  const _Float16* aSrc = Wh + (size_t)(coutBase + am) * KTOT + ak;

  // --- B staging: thread t -> k = t&31 fixed, 16 contiguous cols of row nb ---
  const int kfix = t & 31;
  const int nb   = t >> 5;                  // tile row 0..7
  const int oyTh = oyBase + nb;
  int ci = kfix / 9;                        // decode of K = ci*9 + rr
  int rr = kfix % 9;                        // advances +32 per chunk: rr=(rr+5)%9

  // register staging buffers (pipeline)
  v8h a0, a1;
  _Float16 bst[16];

  // ---- prologue: stage chunk kc=0 into registers ----
  {
    a0 = *(const v8h*)(aSrc);
    a1 = *(const v8h*)(aSrc + 8);
    const int kh = (rr >= 6) ? 2 : (rr >= 3 ? 1 : 0);
    const int kw = rr - 3 * kh;
    const int iy = oyTh + kh - 1;
    const bool rowOK    = (unsigned)iy < (unsigned)HOUT;
    const bool leftBad  = (oxBase == 0)   && (kw == 0);
    const bool rightBad = (oxBase == 112) && (kw == 2);
    if (rowOK) {
      const _Float16* base =
          hB + ((size_t)ci * HOUT + iy) * HOUT + (oxBase + kw - 1);
      bst[0] = leftBad ? (_Float16)0.0f : base[0];
      #pragma unroll
      for (int i = 1; i < 15; ++i) bst[i] = base[i];
      bst[15] = rightBad ? (_Float16)0.0f : base[15];
    } else {
      #pragma unroll
      for (int i = 0; i < 16; ++i) bst[i] = (_Float16)0.0f;
    }
    rr += 5; if (rr >= 9) { rr -= 9; ci += 4; } else ci += 3;
  }

  int buf = 0;
  for (int kc = 0; kc < KTOT; kc += KCH, buf ^= 1) {
    _Float16* Atb = At + buf * ASZ;
    _Float16* Btb = Bt + buf * BSZ;

    // ---- commit staged registers to LDS (buffer `buf`) ----
    *(v8h*)&Atb[am * LDSP + ak]     = a0;
    *(v8h*)&Atb[am * LDSP + ak + 8] = a1;
    #pragma unroll
    for (int i = 0; i < 16; ++i)
      Btb[(nb * 16 + i) * LDSP + kfix] = bst[i];
    // Single barrier: publishes chunk kc AND (since each wave's compute of
    // chunk kc-1 precedes this store in program order) guarantees the other
    // buffer is no longer being read when it is overwritten next iteration.
    __syncthreads();

    // ---- stage NEXT chunk into registers (hidden under WMMAs below) ----
    if (kc + KCH < KTOT) {
      a0 = *(const v8h*)(aSrc + kc + KCH);
      a1 = *(const v8h*)(aSrc + kc + KCH + 8);
      if (kc + 2 * KCH < KTOT)
        __builtin_prefetch(aSrc + kc + 2 * KCH, 0, 1);  // global_prefetch_b8
      const int kh = (rr >= 6) ? 2 : (rr >= 3 ? 1 : 0);
      const int kw = rr - 3 * kh;
      const int iy = oyTh + kh - 1;
      const bool rowOK    = (unsigned)iy < (unsigned)HOUT;
      const bool leftBad  = (oxBase == 0)   && (kw == 0);
      const bool rightBad = (oxBase == 112) && (kw == 2);
      if (rowOK) {
        const _Float16* base =
            hB + ((size_t)ci * HOUT + iy) * HOUT + (oxBase + kw - 1);
        bst[0] = leftBad ? (_Float16)0.0f : base[0];
        #pragma unroll
        for (int i = 1; i < 15; ++i) bst[i] = base[i];
        bst[15] = rightBad ? (_Float16)0.0f : base[15];
      } else {
        #pragma unroll
        for (int i = 0; i < 16; ++i) bst[i] = (_Float16)0.0f;
      }
      rr += 5; if (rr >= 9) { rr -= 9; ci += 4; } else ci += 3;
    }

    // ---- fragments: load A + ALL 8 B fragments, then 8 WMMAs ----
    union { v16h v; v8h p[2]; } afr, bf[8];
    const int mrow = 16 * wave + lhalf;
    // A 16x32 f16 layout: lanes0-15 K0-7 (v0-3)/K16-23 (v4-7); lanes16-31 +8
    afr.p[0] = *(const v8h*)&Atb[mrow * LDSP + hi * 8];
    afr.p[1] = *(const v8h*)&Atb[mrow * LDSP + 16 + hi * 8];
    #pragma unroll
    for (int nf = 0; nf < 8; ++nf) {
      const int ncol = 16 * nf + lhalf;
      // B 32x16 f16 layout: lanes0-15 K0-7/K16-23 (N=lane); lanes16-31 +8
      bf[nf].p[0] = *(const v8h*)&Btb[ncol * LDSP + hi * 8];
      bf[nf].p[1] = *(const v8h*)&Btb[ncol * LDSP + 16 + hi * 8];
    }
    #pragma unroll
    for (int nf = 0; nf < 8; ++nf) {
      acc[nf] = __builtin_amdgcn_wmma_f32_16x16x32_f16(
          false, afr.v, false, bf[nf].v, (short)0, acc[nf], false, false);
    }

    // ---- enforce ds_load/wmma pipeline (lookahead-2) for this region ----
    // 0x100 = DS read, 0x020 = VMEM read, 0x008 = MFMA/WMMA
    SCHED_GB(0x100, 6, 0);          // A frag + B frags 0,1
    SCHED_GB(0x020, 19, 0);         // next-chunk global staging issued early
    SCHED_GB(0x100, 2, 0); SCHED_GB(0x008, 1, 0);   // load bf2 | wmma0
    SCHED_GB(0x100, 2, 0); SCHED_GB(0x008, 1, 0);   // load bf3 | wmma1
    SCHED_GB(0x100, 2, 0); SCHED_GB(0x008, 1, 0);   // load bf4 | wmma2
    SCHED_GB(0x100, 2, 0); SCHED_GB(0x008, 1, 0);   // load bf5 | wmma3
    SCHED_GB(0x100, 2, 0); SCHED_GB(0x008, 1, 0);   // load bf6 | wmma4
    SCHED_GB(0x100, 2, 0); SCHED_GB(0x008, 1, 0);   // load bf7 | wmma5
    SCHED_GB(0x008, 2, 0);                          // wmma6, wmma7
  }

  // ---- epilogue: bias + store (C layout: VGPR r -> M=r+8*hi, N=lhalf) ----
  const int coW = coutBase + 16 * wave + 8 * hi;
  float bv[8];
  #pragma unroll
  for (int r = 0; r < 8; ++r) bv[r] = bias[coW + r];
  #pragma unroll
  for (int nf = 0; nf < 8; ++nf) {
    const int ooy = oyBase + nf;
    const int oox = oxBase + lhalf;
    #pragma unroll
    for (int r = 0; r < 8; ++r) {
      const int co = coW + r;
      y[(((size_t)batch * C_ + co) * HOUT + ooy) * HOUT + oox] = acc[nf][r] + bv[r];
    }
  }
}

// ---------------------------------------------------------------------------
extern "C" void kernel_launch(void* const* d_in, const int* in_sizes, int n_in,
                              void* d_out, int out_size, void* d_ws, size_t ws_size,
                              hipStream_t stream) {
  const float* x = (const float*)d_in[0];   // [16,256,64,64]
  const float* W = (const float*)d_in[1];   // [256,256,3,3]
  const float* b = (const float*)d_in[2];   // [256]
  float* y = (float*)d_out;                 // [16,256,128,128]

  const size_t hElems = (size_t)B_ * C_ * HOUT * HOUT;          // 67,108,864
  _Float16* h  = (_Float16*)d_ws;
  _Float16* Wh = (_Float16*)((char*)d_ws + hElems * sizeof(_Float16));

  // Stage 1: FIR upsample -> h (f16)
  upfir_kernel<<<(unsigned)(hElems / 256), 256, 0, stream>>>(x, h);

  // Stage 2: weights -> f16
  const int wN = C_ * KTOT;                                     // 589,824
  wconv_kernel<<<(wN + 255) / 256, 256, 0, stream>>>(W, Wh, wN);

  // Stage 3: WMMA implicit-GEMM conv
  dim3 grid(HOUT / NT_C, HOUT / NT_R, B_ * (C_ / MTILE));       // (8,16,32)
  conv3x3_wmma_kernel<<<grid, 256, 0, stream>>>(h, Wh, b, y);
}
